// RandlaNet_37022618091670
// MI455X (gfx1250) — matrix operand order repaired and checked by reading, not compiled, not run on hardware
//
#include <hip/hip_runtime.h>
#include <math.h>

// ---------------- problem constants ----------------
constexpr int Bn   = 8;
constexpr int Np   = 4096;   // N points
constexpr int S    = 1024;   // NPOINT
constexpr int NSm  = 32;     // NSAMPLE
constexpr int CIN  = 29;
constexpr int DIN  = 32;     // 3 + CIN
constexpr int NC   = Bn * S * NSm;   // 262144 GEMM columns

typedef __attribute__((ext_vector_type(16))) _Float16 v16h;
typedef __attribute__((ext_vector_type(8)))  float    v8f;

union FragA { v16h v; unsigned int u[8]; };
union FragB { v16h v; uint4 q[2]; };

// ---------------- FPS: one block per batch, dist array in LDS ----------------
__global__ __launch_bounds__(256) void fps_kernel(const float* __restrict__ xyz,
                                                  int* __restrict__ ct_idx) {
  const int b = blockIdx.x;
  const float* xb = xyz + (size_t)b * 3 * Np;
  __shared__ float dist[Np];
  __shared__ float wbest[8];
  __shared__ int   widx[8];
  __shared__ int   sfar;
  const int tid = threadIdx.x;
  for (int i = tid; i < Np; i += 256) dist[i] = 1e10f;
  __syncthreads();
  int far = 0;
  for (int it = 0; it < S; ++it) {
    if (tid == 0) ct_idx[b * S + it] = far;
    const float cx = xb[far], cy = xb[Np + far], cz = xb[2 * Np + far];
    float best = -1.0f; int bi = 0;
    for (int i = tid; i < Np; i += 256) {
      float dx = xb[i] - cx, dy = xb[Np + i] - cy, dz = xb[2 * Np + i] - cz;
      float d  = dx * dx + dy * dy + dz * dz;
      float nd = fminf(dist[i], d);
      dist[i]  = nd;
      if (nd > best) { best = nd; bi = i; }   // strict > keeps first (smallest index)
    }
    // wave32 argmax reduce, tie -> smaller index
    for (int o = 16; o > 0; o >>= 1) {
      float ob = __shfl_down(best, o);
      int   oi = __shfl_down(bi, o);
      if (ob > best || (ob == best && oi < bi)) { best = ob; bi = oi; }
    }
    if ((tid & 31) == 0) { wbest[tid >> 5] = best; widx[tid >> 5] = bi; }
    __syncthreads();
    if (tid == 0) {
      float gb = wbest[0]; int gi = widx[0];
      for (int w = 1; w < 8; ++w)
        if (wbest[w] > gb || (wbest[w] == gb && widx[w] < gi)) { gb = wbest[w]; gi = widx[w]; }
      sfar = gi;
    }
    __syncthreads();
    far = sfar;
  }
}

// ---------------- kNN: one block per centroid, 32 argmin extractions ----------------
__global__ __launch_bounds__(256) void knn_kernel(const float* __restrict__ xyz,
                                                  const int* __restrict__ ct_idx,
                                                  int* __restrict__ gidx) {
  const int bs = blockIdx.x;            // b*S + s
  const int b = bs / S;
  const float* xb = xyz + (size_t)b * 3 * Np;
  __shared__ float d2[Np];
  __shared__ float wbest[8];
  __shared__ int   widx[8];
  __shared__ int   ssel;
  const int tid = threadIdx.x;
  const int ci = ct_idx[bs];
  const float cx = xb[ci], cy = xb[Np + ci], cz = xb[2 * Np + ci];
  for (int i = tid; i < Np; i += 256) {
    float dx = xb[i] - cx, dy = xb[Np + i] - cy, dz = xb[2 * Np + i] - cz;
    d2[i] = dx * dx + dy * dy + dz * dz;
  }
  __syncthreads();
  for (int k = 0; k < NSm; ++k) {
    float best = 3.4e38f; int bi = 0;
    for (int i = tid; i < Np; i += 256) {
      float d = d2[i];
      if (d < best) { best = d; bi = i; }
    }
    for (int o = 16; o > 0; o >>= 1) {
      float ob = __shfl_down(best, o);
      int   oi = __shfl_down(bi, o);
      if (ob < best || (ob == best && oi < bi)) { best = ob; bi = oi; }
    }
    if ((tid & 31) == 0) { wbest[tid >> 5] = best; widx[tid >> 5] = bi; }
    __syncthreads();
    if (tid == 0) {
      float gb = wbest[0]; int gi = widx[0];
      for (int w = 1; w < 8; ++w)
        if (wbest[w] < gb || (wbest[w] == gb && widx[w] < gi)) { gb = wbest[w]; gi = widx[w]; }
      gidx[(size_t)bs * NSm + k] = gi;
      d2[gi] = 3.4e38f;
      ssel = gi;
    }
    __syncthreads();
    (void)ssel;
  }
}

// ---------------- gather: build f16 col-major feature matrices + ct_p output ----------------
__global__ __launch_bounds__(256) void gather_kernel(const float* __restrict__ xyz,
                                                     const float* __restrict__ points,
                                                     const int* __restrict__ ct_idx,
                                                     const int* __restrict__ gidx,
                                                     _Float16* __restrict__ X0,
                                                     _Float16* __restrict__ Xpe,
                                                     float* __restrict__ out_ctp) {
  const int col = blockIdx.x * 256 + threadIdx.x;   // < NC
  const int j  = col & (NSm - 1);
  const int bs = col / NSm;
  const int s  = bs & (S - 1);
  const int b  = bs / S;
  const float* xb = xyz + (size_t)b * 3 * Np;
  const int ci = ct_idx[bs];
  const float cx = xb[ci], cy = xb[Np + ci], cz = xb[2 * Np + ci];
  const int g = gidx[col];
  const float rx = xb[g] - cx, ry = xb[Np + g] - cy, rz = xb[2 * Np + g] - cz;

  _Float16* xc = X0 + (size_t)col * DIN;
  xc[0] = (_Float16)rx; xc[1] = (_Float16)ry; xc[2] = (_Float16)rz;
  const float* pb = points + (size_t)b * CIN * Np;
  #pragma unroll 4
  for (int c = 0; c < CIN; ++c) xc[3 + c] = (_Float16)pb[(size_t)c * Np + g];

  const float rd = sqrtf(fmaxf(rx * rx + ry * ry + rz * rz, 1e-12f));
  _Float16* pc = Xpe + (size_t)col * DIN;
  pc[0] = (_Float16)rx; pc[1] = (_Float16)ry; pc[2] = (_Float16)rz; pc[3] = (_Float16)rd;
  #pragma unroll
  for (int c = 4; c < DIN; ++c) pc[c] = (_Float16)0.0f;

  if (j == 0) {
    out_ctp[((size_t)b * 3 + 0) * S + s] = cx;
    out_ctp[((size_t)b * 3 + 1) * S + s] = cy;
    out_ctp[((size_t)b * 3 + 2) * S + s] = cz;
  }
}

// ---------------- weight prep: f32 -> f16 (conv1 padded 4 -> 32 K) ----------------
__global__ __launch_bounds__(256) void prep_weights_kernel(const float* w0, const float* w1,
                                                           const float* w2, const float* c1,
                                                           const float* c2,
                                                           _Float16* W0h, _Float16* W1h,
                                                           _Float16* W2h, _Float16* C1h,
                                                           _Float16* C2h) {
  const int i = blockIdx.x * 256 + threadIdx.x;
  if (i < 64 * 32)  W0h[i] = (_Float16)w0[i];
  if (i < 64 * 64)  W1h[i] = (_Float16)w1[i];
  if (i < 128 * 64) W2h[i] = (_Float16)w2[i];
  if (i < 64 * 32) {
    const int r = i >> 5, k = i & 31;
    C1h[i] = (_Float16)(k < 4 ? c1[r * 4 + k] : 0.0f);
    C2h[i] = (_Float16)c2[i];
  }
}

// ---------------- WMMA GEMM: Y[M,NC] = W[M,K] * X[K,NC] (col-major X/Y, row-major W) ----------------
__global__ __launch_bounds__(256) void gemm_wmma_kernel(const _Float16* __restrict__ W,
                                                        const _Float16* __restrict__ X,
                                                        float* __restrict__ Y,
                                                        int M, int K) {
  const int lane  = threadIdx.x & 31;
  const int wave  = threadIdx.x >> 5;
  const int ntile = blockIdx.x * 8 + wave;
  const int mtile = blockIdx.y;
  const int half  = lane >> 4;          // 0 or 1
  const int idx16 = lane & 15;

  const int rowA = mtile * 16 + idx16;
  const int col  = ntile * 16 + idx16;
  const _Float16* wrow = W + (size_t)rowA * K;
  const _Float16* xcol = X + (size_t)col  * K;

  v8f acc = {};
  for (int kk = 0; kk < K; kk += 32) {
    FragA a; FragB bf;
    const int h8 = half * 8;
    #pragma unroll
    for (int j = 0; j < 4; ++j) {
      a.u[j]     = *(const unsigned int*)(wrow + kk + h8 + 2 * j);        // K = h8+{0,2,4,6}
      a.u[4 + j] = *(const unsigned int*)(wrow + kk + 16 + h8 + 2 * j);   // K = 16+h8+{0..6}
    }
    const uint4* bp = (const uint4*)(xcol + kk + h8);                     // K = h8+{0..7}, 16+h8+{0..7}
    bf.q[0] = bp[0];
    bf.q[1] = bp[2];
    acc = __builtin_amdgcn_wmma_f32_16x16x32_f16(false, a.v, false, bf.v,
                                                 (short)0, acc, false, false);
  }
  float* ycol = Y + (size_t)col * M;
  const int r0 = mtile * 16 + half * 8;
  #pragma unroll
  for (int r = 0; r < 8; ++r) ycol[r0 + r] = acc[r];
}

// ---------------- BN helpers ----------------
__global__ __launch_bounds__(256) void zero_kernel(float* p) { p[threadIdx.x] = 0.0f; }

__global__ __launch_bounds__(256) void bn_stats_kernel(const float* __restrict__ Y, int M,
                                                       float* __restrict__ sums) {
  const int c = blockIdx.y;
  __shared__ float sa[256], sb[256];
  float a = 0.0f, q = 0.0f;
  for (int n = blockIdx.x * 256 + threadIdx.x; n < NC; n += gridDim.x * 256) {
    const float v = Y[(size_t)n * M + c];
    a += v; q += v * v;
  }
  sa[threadIdx.x] = a; sb[threadIdx.x] = q;
  __syncthreads();
  for (int st = 128; st > 0; st >>= 1) {
    if (threadIdx.x < st) { sa[threadIdx.x] += sa[threadIdx.x + st]; sb[threadIdx.x] += sb[threadIdx.x + st]; }
    __syncthreads();
  }
  if (threadIdx.x == 0) {
    atomicAdd(&sums[c], sa[0]);
    atomicAdd(&sums[M + c], sb[0]);
  }
}

__global__ __launch_bounds__(256) void bn_apply_kernel(const float* __restrict__ Y,
                                                       const float* __restrict__ sums,
                                                       const float* __restrict__ gamma,
                                                       const float* __restrict__ beta,
                                                       int M,
                                                       _Float16* __restrict__ outH,
                                                       float* __restrict__ outF) {
  const size_t i = (size_t)blockIdx.x * 256 + threadIdx.x;
  if (i >= (size_t)M * NC) return;
  const int c = (int)(i % M);
  const float inv = 1.0f / (float)NC;
  const float mean = sums[c] * inv;
  const float var  = sums[M + c] * inv - mean * mean;
  const float sc   = gamma[c] * rsqrtf(var + 1e-5f);
  const float v    = fmaxf(sc * (Y[i] - mean) + beta[c], 0.0f);
  if (outH) outH[i] = (_Float16)v;
  if (outF) outF[i] = v;
}

// ---------------- pooling ----------------
__global__ __launch_bounds__(256) void max_ns_kernel(const float* __restrict__ act,  // 128 x NC col-major
                                                     float* __restrict__ out) {      // conv_points base
  const int i = blockIdx.x * 256 + threadIdx.x;     // B*128*S
  if (i >= Bn * 128 * S) return;
  const int s = i & (S - 1);
  const int t = i / S;
  const int c = t & 127;
  const int b = t >> 7;
  const size_t base = ((size_t)(b * S + s)) * NSm;
  float m = -3.4e38f;
  #pragma unroll 4
  for (int j = 0; j < NSm; ++j) m = fmaxf(m, act[(base + j) * 128 + c]);
  out[((size_t)b * 384 + c) * S + s] = m;
}

__global__ __launch_bounds__(256) void meanmax_kernel(const float* __restrict__ F1,  // 64 x NC
                                                      const float* __restrict__ F2,  // 64 x NC
                                                      float* __restrict__ out) {
  const int i = blockIdx.x * 256 + threadIdx.x;     // B*128*S
  if (i >= Bn * 128 * S) return;
  const int s = i & (S - 1);
  const int t = i / S;
  const int c = t & 127;
  const int b = t >> 7;
  const float* F = (c < 64) ? F1 : F2;
  const int cc = c & 63;
  const size_t base = ((size_t)(b * S + s)) * NSm;
  float m = -3.4e38f, sum = 0.0f;
  #pragma unroll 4
  for (int j = 0; j < NSm; ++j) {
    const float v = F[(base + j) * 64 + cc];
    sum += v; m = fmaxf(m, v);
  }
  out[((size_t)b * 384 + 128 + c) * S + s] = sum * (1.0f / NSm);
  out[((size_t)b * 384 + 256 + c) * S + s] = m;
}

// ---------------- host launch ----------------
extern "C" void kernel_launch(void* const* d_in, const int* in_sizes, int n_in,
                              void* d_out, int out_size, void* d_ws, size_t ws_size,
                              hipStream_t stream) {
  const float* xyz    = (const float*)d_in[0];
  const float* points = (const float*)d_in[1];
  const float* w0 = (const float*)d_in[2];
  const float* g0 = (const float*)d_in[4];
  const float* be0 = (const float*)d_in[5];
  const float* w1 = (const float*)d_in[6];
  const float* g1 = (const float*)d_in[8];
  const float* be1 = (const float*)d_in[9];
  const float* w2 = (const float*)d_in[10];
  const float* g2 = (const float*)d_in[12];
  const float* be2 = (const float*)d_in[13];
  const float* c1w = (const float*)d_in[14];
  const float* bn1g = (const float*)d_in[16];
  const float* bn1be = (const float*)d_in[17];
  const float* c2w = (const float*)d_in[18];
  const float* bn2g = (const float*)d_in[20];
  const float* bn2be = (const float*)d_in[21];
  float* out = (float*)d_out;

  char* ws = (char*)d_ws;
  size_t off = 0;
  auto carve = [&](size_t bytes) -> void* {
    void* p = ws + off;
    off += (bytes + 255) & ~(size_t)255;
    return p;
  };
  int*       ct_idx = (int*)carve((size_t)Bn * S * 4);
  int*       gidx   = (int*)carve((size_t)NC * 4);
  _Float16*  X0     = (_Float16*)carve((size_t)NC * DIN * 2);
  _Float16*  Xpe    = (_Float16*)carve((size_t)NC * DIN * 2);
  _Float16*  Xh     = (_Float16*)carve((size_t)NC * 64 * 2);
  float*     bufA   = (float*)carve((size_t)NC * 128 * 4);
  _Float16*  W0h    = (_Float16*)carve(64 * 32 * 2);
  _Float16*  W1h    = (_Float16*)carve(64 * 64 * 2);
  _Float16*  W2h    = (_Float16*)carve(128 * 64 * 2);
  _Float16*  C1h    = (_Float16*)carve(64 * 32 * 2);
  _Float16*  C2h    = (_Float16*)carve(64 * 32 * 2);
  float*     statsBuf = (float*)carve(256 * 4);
  (void)ws_size; (void)n_in; (void)in_sizes; (void)out_size;

  prep_weights_kernel<<<32, 256, 0, stream>>>(w0, w1, w2, c1w, c2w, W0h, W1h, W2h, C1h, C2h);
  fps_kernel<<<Bn, 256, 0, stream>>>(xyz, ct_idx);
  knn_kernel<<<Bn * S, 256, 0, stream>>>(xyz, ct_idx, gidx);
  gather_kernel<<<NC / 256, 256, 0, stream>>>(xyz, points, ct_idx, gidx, X0, Xpe, out);

  const int nblk = NC / 128;                     // GEMM N-tile blocks (8 waves per block)
  const int eblk64  = (64 * NC) / 256;
  const int eblk128 = (128 * NC) / 256;

  // ---- MLP layer 0: 64x32 ----
  zero_kernel<<<1, 256, 0, stream>>>(statsBuf);
  gemm_wmma_kernel<<<dim3(nblk, 4), 256, 0, stream>>>(W0h, X0, bufA, 64, 32);
  bn_stats_kernel<<<dim3(64, 64), 256, 0, stream>>>(bufA, 64, statsBuf);
  bn_apply_kernel<<<eblk64, 256, 0, stream>>>(bufA, statsBuf, g0, be0, 64, Xh, nullptr);

  // ---- MLP layer 1: 64x64 ----
  zero_kernel<<<1, 256, 0, stream>>>(statsBuf);
  gemm_wmma_kernel<<<dim3(nblk, 4), 256, 0, stream>>>(W1h, Xh, bufA, 64, 64);
  bn_stats_kernel<<<dim3(64, 64), 256, 0, stream>>>(bufA, 64, statsBuf);
  bn_apply_kernel<<<eblk64, 256, 0, stream>>>(bufA, statsBuf, g1, be1, 64, Xh, nullptr);

  // ---- MLP layer 2: 128x64 ----
  zero_kernel<<<1, 256, 0, stream>>>(statsBuf);
  gemm_wmma_kernel<<<dim3(nblk, 8), 256, 0, stream>>>(W2h, Xh, bufA, 128, 64);
  bn_stats_kernel<<<dim3(64, 128), 256, 0, stream>>>(bufA, 128, statsBuf);
  bn_apply_kernel<<<eblk128, 256, 0, stream>>>(bufA, statsBuf, g2, be2, 128, nullptr, bufA);
  max_ns_kernel<<<(Bn * 128 * S) / 256, 256, 0, stream>>>(bufA, out + (size_t)Bn * 3 * S);

  // ---- PE branch: conv1 on [rel,rd] and conv2 on ft, both 64x32 ----
  float* Ype1 = bufA;
  float* Ype2 = bufA + (size_t)64 * NC;
  zero_kernel<<<1, 256, 0, stream>>>(statsBuf);
  gemm_wmma_kernel<<<dim3(nblk, 4), 256, 0, stream>>>(C1h, Xpe, Ype1, 64, 32);
  bn_stats_kernel<<<dim3(64, 64), 256, 0, stream>>>(Ype1, 64, statsBuf);
  bn_apply_kernel<<<eblk64, 256, 0, stream>>>(Ype1, statsBuf, bn1g, bn1be, 64, nullptr, Ype1);

  zero_kernel<<<1, 256, 0, stream>>>(statsBuf);
  gemm_wmma_kernel<<<dim3(nblk, 4), 256, 0, stream>>>(C2h, X0, Ype2, 64, 32);
  bn_stats_kernel<<<dim3(64, 64), 256, 0, stream>>>(Ype2, 64, statsBuf);
  bn_apply_kernel<<<eblk64, 256, 0, stream>>>(Ype2, statsBuf, bn2g, bn2be, 64, nullptr, Ype2);

  meanmax_kernel<<<(Bn * 128 * S) / 256, 256, 0, stream>>>(Ype1, Ype2, out + (size_t)Bn * 3 * S);
}
